// SpatialTransformer_27023934226514
// MI455X (gfx1250) — matrix operand order repaired and checked by reading, not compiled, not run on hardware
//
#include <hip/hip_runtime.h>

// ---------------------------------------------------------------------------
// Types / helpers
// ---------------------------------------------------------------------------
typedef _Float16 half_t;
typedef __attribute__((ext_vector_type(16))) _Float16 v16h;
typedef __attribute__((ext_vector_type(8)))  float    v8f;

union frag16 { v16h h; uint4 u[2]; };

__device__ __forceinline__ v8f wmma_f16(v16h a, v16h b, v8f c) {
  // D = A(16x32 f16) * B(32x16 f16) + C(16x16 f32)
  return __builtin_amdgcn_wmma_f32_16x16x32_f16(
      /*neg_a=*/false, a, /*neg_b=*/false, b,
      /*c_mod=*/(short)0, c, /*reuse_a=*/false, /*reuse_b=*/false);
}

__device__ __forceinline__ int imin(int a, int b) { return a < b ? a : b; }

static constexpr int Cc   = 512;
static constexpr int Pp   = 2304;   // H*W
static constexpr int Bn   = 2;
static constexpr int M0c  = Bn * Pp;  // 4608
static constexpr int Tt   = 77;
static constexpr int MTc  = Bn * Tt;  // 154
static constexpr int Ff   = 2048;     // 4*C
static constexpr int NHc  = 8;
static constexpr int HDc  = 64;
static constexpr int SKPX = 96;       // cross-attn padded key stride
static constexpr float EPSc = 1e-5f;
static constexpr float SCALEc = 8.0f; // sqrt(C/NH), q is MULTIPLIED (faithful)

// ---------------------------------------------------------------------------
// Block-wide sum of (s, ss) — returns totals to all threads (256-thread block)
// ---------------------------------------------------------------------------
__device__ __forceinline__ float2 block_sum2(float s, float ss) {
  __shared__ float rs[8], rss[8];
  for (int m = 1; m < 32; m <<= 1) {
    s  += __shfl_xor(s,  m, 32);
    ss += __shfl_xor(ss, m, 32);
  }
  int wid = threadIdx.x >> 5, lane = threadIdx.x & 31;
  if (lane == 0) { rs[wid] = s; rss[wid] = ss; }
  __syncthreads();
  float a = 0.f, b = 0.f;
  if (threadIdx.x < 8) { a = rs[threadIdx.x]; b = rss[threadIdx.x]; }
  for (int m = 1; m < 8; m <<= 1) {
    a += __shfl_xor(a, m, 8);
    b += __shfl_xor(b, m, 8);
  }
  if (threadIdx.x == 0) { rs[0] = a; rss[0] = b; }
  __syncthreads();
  return make_float2(rs[0], rss[0]);
}

// ---------------------------------------------------------------------------
// Weight transpose + f32->f16:  wt[n*K + k] = w[k*N + n]
// ---------------------------------------------------------------------------
__global__ __launch_bounds__(256)
void wtrans_kernel(const float* __restrict__ w, half_t* __restrict__ wt,
                   int K, int N) {
  size_t i = (size_t)blockIdx.x * 256 + threadIdx.x;
  if (i >= (size_t)K * N) return;
  int n = (int)(i % N);
  int k = (int)(i / N);
  wt[(size_t)n * K + k] = (half_t)w[i];
}

// f32 -> f16 elementwise
__global__ __launch_bounds__(256)
void f32tof16_kernel(const float* __restrict__ a, half_t* __restrict__ o, size_t n) {
  size_t i = (size_t)blockIdx.x * 256 + threadIdx.x;
  if (i < n) o[i] = (half_t)a[i];
}

// ---------------------------------------------------------------------------
// GroupNorm: x[B,C,H,W] -> out[B,HW,C] (f16), 32 groups of 16 channels
// grid = B*32 blocks
// ---------------------------------------------------------------------------
__global__ __launch_bounds__(256)
void groupnorm_kernel(const float* __restrict__ x, const float* __restrict__ g,
                      const float* __restrict__ bt, half_t* __restrict__ out) {
  const int CPG = 16;
  int b  = blockIdx.x >> 5;
  int gr = blockIdx.x & 31;
  const float* xb = x + ((size_t)b * Cc + gr * CPG) * Pp;
  float s = 0.f, ss = 0.f;
  for (int i = threadIdx.x; i < CPG * Pp; i += 256) {
    float v = xb[i];
    s += v; ss += v * v;
  }
  float2 tot = block_sum2(s, ss);
  const float inv_n = 1.0f / (float)(CPG * Pp);
  float mean = tot.x * inv_n;
  float var  = tot.y * inv_n - mean * mean;
  float rinv = rsqrtf(var + EPSc);
  for (int i = threadIdx.x; i < CPG * Pp; i += 256) {
    int c = gr * CPG + i / Pp;
    int p = i % Pp;
    float v = (xb[i] - mean) * rinv * g[c] + bt[c];
    out[((size_t)b * Pp + p) * Cc + c] = (half_t)v;
  }
}

// ---------------------------------------------------------------------------
// LayerNorm over last dim (C=512): in f32 [M,C] -> out f16 [M,C]; grid = M
// ---------------------------------------------------------------------------
__global__ __launch_bounds__(256)
void layernorm_kernel(const float* __restrict__ x, const float* __restrict__ g,
                      const float* __restrict__ bt, half_t* __restrict__ out) {
  size_t row = blockIdx.x;
  const float* xr = x + row * Cc;
  float x0 = xr[threadIdx.x], x1 = xr[threadIdx.x + 256];
  float2 tot = block_sum2(x0 + x1, x0 * x0 + x1 * x1);
  float mean = tot.x * (1.0f / Cc);
  float var  = tot.y * (1.0f / Cc) - mean * mean;
  float rinv = rsqrtf(var + EPSc);
  half_t* orow = out + row * Cc;
  orow[threadIdx.x]       = (half_t)((x0 - mean) * rinv * g[threadIdx.x] + bt[threadIdx.x]);
  orow[threadIdx.x + 256] = (half_t)((x1 - mean) * rinv * g[threadIdx.x + 256] + bt[threadIdx.x + 256]);
}

// ---------------------------------------------------------------------------
// V transpose for attention: vt[(bh*64 + d)*skp + j] = v[(b*seqk + j)*C + h*64 + d]
// zero-pads j in [seqk, skp)
// ---------------------------------------------------------------------------
__global__ __launch_bounds__(256)
void vtrans_kernel(const half_t* __restrict__ v, half_t* __restrict__ vt,
                   int seqk, int skp) {
  size_t i = (size_t)blockIdx.x * 256 + threadIdx.x;
  size_t total = (size_t)Bn * NHc * HDc * skp;
  if (i >= total) return;
  int j  = (int)(i % skp);
  int d  = (int)((i / skp) % HDc);
  int bh = (int)(i / ((size_t)skp * HDc));
  int b = bh >> 3, h = bh & 7;
  half_t val = (half_t)0.0f;
  if (j < seqk) val = v[((size_t)b * seqk + j) * Cc + h * HDc + d];
  vt[i] = val;
}

// ---------------------------------------------------------------------------
// WMMA GEMM: out = (A[M,K](f16) @ W[K,N] + bias) * scale (+ residual)
// Bt is W transposed: [N,K] f16.
// Wave tile 32(M) x 64(N): 2 A frags x 4 B frags -> 8 WMMAs per k-step,
// ~21 FLOP/byte from memory (vs 8 for a 16x16 tile).
// Block = 8 waves as 2(M) x 4(N) -> 64 x 256 block tile. N must be mult of 64;
// all N used here are multiples of 256.
// ---------------------------------------------------------------------------
__global__ __launch_bounds__(256)
void gemm_wmma_kernel(const half_t* __restrict__ A, const half_t* __restrict__ Bt,
                      const float* __restrict__ bias, const float* __restrict__ resid,
                      float* __restrict__ outF, half_t* __restrict__ outH,
                      int M, int N, int K, float scale) {
  const int lane = threadIdx.x & 31;
  const int wid  = threadIdx.x >> 5;
  const int wm = wid >> 2, wn = wid & 3;
  const int row0 = blockIdx.y * 64 + wm * 32;
  const int col0 = blockIdx.x * 256 + wn * 64;
  if (row0 >= M || col0 >= N) return;         // wave-uniform; EXEC stays full
  const int l15 = lane & 15;
  const int hb  = lane >> 4;

  const int arow0 = imin(row0 + l15,      M - 1);
  const int arow1 = imin(row0 + 16 + l15, M - 1);
  const half_t* Ap0 = A  + (size_t)arow0 * K + hb * 8;
  const half_t* Ap1 = A  + (size_t)arow1 * K + hb * 8;
  const half_t* Bp  = Bt + (size_t)(col0 + l15) * K + hb * 8;
  const size_t bstep = (size_t)16 * K;        // 16 columns of Bt

  v8f acc0[4] = {{}, {}, {}, {}};
  v8f acc1[4] = {{}, {}, {}, {}};

  #pragma clang loop unroll_count(2)
  for (int k = 0; k < K; k += 32) {
    frag16 a0, a1;
    // A 16x32 f16 fragment: lane row l15, halves [k+hb*8, +8) and [k+hb*8+16, +8)
    a0.u[0] = *(const uint4*)(Ap0 + k);
    a0.u[1] = *(const uint4*)(Ap0 + k + 16);
    a1.u[0] = *(const uint4*)(Ap1 + k);
    a1.u[1] = *(const uint4*)(Ap1 + k + 16);
    #pragma unroll
    for (int j = 0; j < 4; ++j) {
      frag16 b;
      // B 32x16 f16 fragment: lane col l15, halves [k+hb*8, +8) and [k+hb*8+16, +8)
      b.u[0] = *(const uint4*)(Bp + j * bstep + k);
      b.u[1] = *(const uint4*)(Bp + j * bstep + k + 16);
      acc0[j] = wmma_f16(a0.h, b.h, acc0[j]);
      acc1[j] = wmma_f16(a1.h, b.h, acc1[j]);
    }
  }

  float bc[4];
  #pragma unroll
  for (int j = 0; j < 4; ++j)
    bc[j] = bias ? bias[col0 + j * 16 + l15] : 0.0f;

  #pragma unroll
  for (int i = 0; i < 2; ++i) {
    #pragma unroll
    for (int r = 0; r < 8; ++r) {
      int orow = row0 + i * 16 + hb * 8 + r;
      if (orow < M) {
        #pragma unroll
        for (int j = 0; j < 4; ++j) {
          float v = ((i == 0 ? acc0[j][r] : acc1[j][r]) + bc[j]) * scale;
          size_t idx = (size_t)orow * N + col0 + j * 16 + l15;
          if (resid) v += resid[idx];
          if (outF) outF[idx] = v;
          if (outH) outH[idx] = (half_t)v;
        }
      }
    }
  }
}

// ---------------------------------------------------------------------------
// Flash-style attention. One wave = 16 query rows. Online softmax with
// 16-lane butterflies. P re-layout (C-frag -> A-frag) via per-wave LDS.
//   Q : [B*seqq, C] f16 (already scaled), K : [B*seqk, C] f16,
//   Vt: [B*NH, 64, skp] f16, O : [B*seqq, C] f16
// grid = (B*NH, seqq/128), block = 256
// ---------------------------------------------------------------------------
__global__ __launch_bounds__(256)
void attn_wmma_kernel(const half_t* __restrict__ Q, const half_t* __restrict__ Kc,
                      const half_t* __restrict__ Vt, half_t* __restrict__ O,
                      int seqq, int seqk, int skp) {
  __shared__ alignas(16) half_t Plds[8][16][40];
  const int lane = threadIdx.x & 31;
  const int wid  = threadIdx.x >> 5;
  const int bh = blockIdx.x;
  const int b = bh >> 3, h = bh & 7;
  const int qi0 = blockIdx.y * 128 + wid * 16;
  if (qi0 >= seqq) return;
  const int l15 = lane & 15, hb = lane >> 4;

  // Q fragments: d 0..31 (qf0), d 32..63 (qf1)
  const half_t* Qp = Q + ((size_t)b * seqq + qi0 + l15) * Cc + h * HDc + hb * 16;
  frag16 qf0, qf1;
  qf0.u[0] = *(const uint4*)(Qp);
  qf0.u[1] = *(const uint4*)(Qp + 8);
  qf1.u[0] = *(const uint4*)(Qp + 32);
  qf1.u[1] = *(const uint4*)(Qp + 40);

  v8f o0 = {}, o1 = {}, o2 = {}, o3 = {};
  float mrow[8], lrow[8];
  #pragma unroll
  for (int r = 0; r < 8; ++r) { mrow[r] = -1e30f; lrow[r] = 0.0f; }

  for (int j0 = 0; j0 < seqk; j0 += 32) {
    const int keyA = j0 + l15;        // S0 column for this lane
    const int keyB = j0 + 16 + l15;   // S1 column for this lane
    const bool va = keyA < seqk;
    const bool vb = keyB < seqk;
    const half_t* Kpa = Kc + ((size_t)b * seqk + imin(keyA, seqk - 1)) * Cc + h * HDc + hb * 16;
    const half_t* Kpb = Kc + ((size_t)b * seqk + imin(keyB, seqk - 1)) * Cc + h * HDc + hb * 16;

    frag16 kf;
    v8f s0 = {}, s1 = {};
    kf.u[0] = *(const uint4*)(Kpa);       kf.u[1] = *(const uint4*)(Kpa + 8);
    s0 = wmma_f16(qf0.h, kf.h, s0);
    kf.u[0] = *(const uint4*)(Kpa + 32);  kf.u[1] = *(const uint4*)(Kpa + 40);
    s0 = wmma_f16(qf1.h, kf.h, s0);
    kf.u[0] = *(const uint4*)(Kpb);       kf.u[1] = *(const uint4*)(Kpb + 8);
    s1 = wmma_f16(qf0.h, kf.h, s1);
    kf.u[0] = *(const uint4*)(Kpb + 32);  kf.u[1] = *(const uint4*)(Kpb + 40);
    s1 = wmma_f16(qf1.h, kf.h, s1);

    #pragma unroll
    for (int r = 0; r < 8; ++r) {
      // tile row-max across the 16 lanes holding this row
      float v = va ? s0[r] : -1e30f;
      if (vb) v = fmaxf(v, s1[r]);
      for (int m = 1; m < 16; m <<= 1) v = fmaxf(v, __shfl_xor(v, m, 16));
      float mnew = fmaxf(mrow[r], v);
      float corr = __expf(mrow[r] - mnew);
      mrow[r] = mnew;
      lrow[r] *= corr;
      o0[r] *= corr; o1[r] *= corr; o2[r] *= corr; o3[r] *= corr;
      float p0 = va ? __expf(s0[r] - mnew) : 0.0f;
      float p1 = vb ? __expf(s1[r] - mnew) : 0.0f;
      float ps = p0 + p1;
      for (int m = 1; m < 16; m <<= 1) ps += __shfl_xor(ps, m, 16);
      lrow[r] += ps;
      // stage P into LDS (C-layout -> row-major 16x32 tile)
      Plds[wid][hb * 8 + r][l15]      = (half_t)p0;
      Plds[wid][hb * 8 + r][16 + l15] = (half_t)p1;
    }

    // reload P as an A fragment (16 queries x 32 keys)
    frag16 pf;
    const half_t* Pq = &Plds[wid][l15][hb * 16];
    pf.u[0] = *(const uint4*)(Pq);
    pf.u[1] = *(const uint4*)(Pq + 8);

    // V fragments from Vt [bh*64 + d][skp]: B = V (32 keys x 16 dims)
    frag16 vf;
    const half_t* V0 = Vt + ((size_t)bh * HDc + l15) * skp + j0 + hb * 16;
    const size_t vstep = (size_t)16 * skp;
    vf.u[0] = *(const uint4*)(V0);                 vf.u[1] = *(const uint4*)(V0 + 8);
    o0 = wmma_f16(pf.h, vf.h, o0);
    vf.u[0] = *(const uint4*)(V0 + vstep);         vf.u[1] = *(const uint4*)(V0 + vstep + 8);
    o1 = wmma_f16(pf.h, vf.h, o1);
    vf.u[0] = *(const uint4*)(V0 + 2 * vstep);     vf.u[1] = *(const uint4*)(V0 + 2 * vstep + 8);
    o2 = wmma_f16(pf.h, vf.h, o2);
    vf.u[0] = *(const uint4*)(V0 + 3 * vstep);     vf.u[1] = *(const uint4*)(V0 + 3 * vstep + 8);
    o3 = wmma_f16(pf.h, vf.h, o3);
  }

  #pragma unroll
  for (int r = 0; r < 8; ++r) {
    float inv = lrow[r] > 0.0f ? 1.0f / lrow[r] : 0.0f;
    size_t orow = (size_t)b * seqq + qi0 + hb * 8 + r;
    half_t* Op = O + orow * Cc + h * HDc;
    Op[l15]      = (half_t)(o0[r] * inv);
    Op[16 + l15] = (half_t)(o1[r] * inv);
    Op[32 + l15] = (half_t)(o2[r] * inv);
    Op[48 + l15] = (half_t)(o3[r] * inv);
  }
}

// ---------------------------------------------------------------------------
// Final: d_out[b,c,p] = proj[b*P+p, c] + x_in[b,c,p]
// ---------------------------------------------------------------------------
__global__ __launch_bounds__(256)
void final_add_kernel(const float* __restrict__ proj, const float* __restrict__ xin,
                      float* __restrict__ out) {
  size_t i = (size_t)blockIdx.x * 256 + threadIdx.x;
  size_t total = (size_t)Bn * Cc * Pp;
  if (i >= total) return;
  int p = (int)(i % Pp);
  int c = (int)((i / Pp) % Cc);
  int b = (int)(i / ((size_t)Pp * Cc));
  out[i] = proj[((size_t)b * Pp + p) * Cc + c] + xin[i];
}

// ---------------------------------------------------------------------------
// Host side
// ---------------------------------------------------------------------------
static void launch_gemm(const half_t* A, const half_t* Bt, const float* bias,
                        const float* resid, float* outF, half_t* outH,
                        int M, int N, int K, float scale, hipStream_t s) {
  dim3 grid((N + 255) / 256, (M + 63) / 64);
  gemm_wmma_kernel<<<grid, 256, 0, s>>>(A, Bt, bias, resid, outF, outH, M, N, K, scale);
}

extern "C" void kernel_launch(void* const* d_in, const int* in_sizes, int n_in,
                              void* d_out, int out_size, void* d_ws, size_t ws_size,
                              hipStream_t stream) {
  const float* x      = (const float*)d_in[0];
  const float* cond   = (const float*)d_in[1];
  const float* gn_g   = (const float*)d_in[2];
  const float* gn_b   = (const float*)d_in[3];
  const float* win_w  = (const float*)d_in[4];
  const float* win_b  = (const float*)d_in[5];
  const float* wout_w = (const float*)d_in[6];
  const float* wout_b = (const float*)d_in[7];
  const float* Wq1 = (const float*)d_in[8];  const float* bq1 = (const float*)d_in[9];
  const float* Wk1 = (const float*)d_in[10]; const float* bk1 = (const float*)d_in[11];
  const float* Wv1 = (const float*)d_in[12]; const float* bv1 = (const float*)d_in[13];
  const float* Wo1 = (const float*)d_in[14]; const float* bo1 = (const float*)d_in[15];
  const float* ln1_g = (const float*)d_in[16]; const float* ln1_b = (const float*)d_in[17];
  const float* Wq2 = (const float*)d_in[18]; const float* bq2 = (const float*)d_in[19];
  const float* Wk2 = (const float*)d_in[20]; const float* bk2 = (const float*)d_in[21];
  const float* Wv2 = (const float*)d_in[22]; const float* bv2 = (const float*)d_in[23];
  const float* Wo2 = (const float*)d_in[24]; const float* bo2 = (const float*)d_in[25];
  const float* ln2_g = (const float*)d_in[26]; const float* ln2_b = (const float*)d_in[27];
  const float* fw1 = (const float*)d_in[28]; const float* fb1 = (const float*)d_in[29];
  const float* fw2 = (const float*)d_in[30]; const float* fb2 = (const float*)d_in[31];
  const float* ln3_g = (const float*)d_in[32]; const float* ln3_b = (const float*)d_in[33];
  (void)in_sizes; (void)n_in; (void)out_size; (void)ws_size;

  char* cur = (char*)d_ws;
  auto alloc = [&](size_t n) -> char* {
    char* p = cur; cur += (n + 255) & ~(size_t)255; return p;
  };
  float*  t     = (float*) alloc((size_t)M0c * Cc * 4);
  half_t* tn    = (half_t*)alloc((size_t)M0c * Cc * 2);
  half_t* qh    = (half_t*)alloc((size_t)M0c * Cc * 2);
  half_t* kh    = (half_t*)alloc((size_t)M0c * Cc * 2);
  half_t* vh    = (half_t*)alloc((size_t)M0c * Cc * 2);
  half_t* vt    = (half_t*)alloc((size_t)Bn * NHc * HDc * Pp * 2);
  half_t* ah    = (half_t*)alloc((size_t)M0c * Cc * 2);
  half_t* mid   = (half_t*)alloc((size_t)M0c * Ff * 2);
  half_t* condh = (half_t*)alloc((size_t)MTc * Cc * 2);
  float*  proj  = (float*) alloc((size_t)M0c * Cc * 4);
  const size_t WCC = (size_t)Cc * Cc;
  half_t* win_t  = (half_t*)alloc(WCC * 2);
  half_t* wout_t = (half_t*)alloc(WCC * 2);
  half_t* wq1t = (half_t*)alloc(2 * WCC * 2);
  half_t* wk1t = (half_t*)alloc(2 * WCC * 2);
  half_t* wv1t = (half_t*)alloc(2 * WCC * 2);
  half_t* wo1t = (half_t*)alloc(2 * WCC * 2);
  half_t* wq2t = (half_t*)alloc(2 * WCC * 2);
  half_t* wk2t = (half_t*)alloc(2 * WCC * 2);
  half_t* wv2t = (half_t*)alloc(2 * WCC * 2);
  half_t* wo2t = (half_t*)alloc(2 * WCC * 2);
  half_t* fw1t = (half_t*)alloc((size_t)2 * Cc * Ff * 2);
  half_t* fw2t = (half_t*)alloc((size_t)2 * Ff * Cc * 2);

  auto tw = [&](const float* w, half_t* wt, int K, int N) {
    wtrans_kernel<<<dim3((unsigned)(((size_t)K * N + 255) / 256)), 256, 0, stream>>>(w, wt, K, N);
  };

  // Weight prep (f16, transposed to [N,K])
  tw(win_w,  win_t,  Cc, Cc);
  tw(wout_w, wout_t, Cc, Cc);
  for (int l = 0; l < 2; ++l) {
    tw(Wq1 + l * WCC, wq1t + l * WCC, Cc, Cc);
    tw(Wk1 + l * WCC, wk1t + l * WCC, Cc, Cc);
    tw(Wv1 + l * WCC, wv1t + l * WCC, Cc, Cc);
    tw(Wo1 + l * WCC, wo1t + l * WCC, Cc, Cc);
    tw(Wq2 + l * WCC, wq2t + l * WCC, Cc, Cc);
    tw(Wk2 + l * WCC, wk2t + l * WCC, Cc, Cc);
    tw(Wv2 + l * WCC, wv2t + l * WCC, Cc, Cc);
    tw(Wo2 + l * WCC, wo2t + l * WCC, Cc, Cc);
    tw(fw1 + (size_t)l * Cc * Ff, fw1t + (size_t)l * Ff * Cc, Cc, Ff);
    tw(fw2 + (size_t)l * Ff * Cc, fw2t + (size_t)l * Cc * Ff, Ff, Cc);
  }

  // cond -> f16
  f32tof16_kernel<<<dim3((unsigned)(((size_t)MTc * Cc + 255) / 256)), 256, 0, stream>>>(
      cond, condh, (size_t)MTc * Cc);

  // GroupNorm -> tn (as [B,HW,C] f16), then input projection -> t
  groupnorm_kernel<<<dim3(Bn * 32), 256, 0, stream>>>(x, gn_g, gn_b, tn);
  launch_gemm(tn, win_t, win_b, nullptr, t, nullptr, M0c, Cc, Cc, 1.0f, stream);

  const dim3 attn_grid(Bn * NHc, (Pp + 127) / 128);
  const unsigned vtb_self  = (unsigned)(((size_t)Bn * NHc * HDc * Pp   + 255) / 256);
  const unsigned vtb_cross = (unsigned)(((size_t)Bn * NHc * HDc * SKPX + 255) / 256);

  for (int l = 0; l < 2; ++l) {
    // ---- self-attention ----
    layernorm_kernel<<<dim3(M0c), 256, 0, stream>>>(t, ln1_g + l * Cc, ln1_b + l * Cc, tn);
    launch_gemm(tn, wq1t + l * WCC, bq1 + l * Cc, nullptr, nullptr, qh, M0c, Cc, Cc, SCALEc, stream);
    launch_gemm(tn, wk1t + l * WCC, bk1 + l * Cc, nullptr, nullptr, kh, M0c, Cc, Cc, 1.0f, stream);
    launch_gemm(tn, wv1t + l * WCC, bv1 + l * Cc, nullptr, nullptr, vh, M0c, Cc, Cc, 1.0f, stream);
    vtrans_kernel<<<dim3(vtb_self), 256, 0, stream>>>(vh, vt, Pp, Pp);
    attn_wmma_kernel<<<attn_grid, 256, 0, stream>>>(qh, kh, vt, ah, Pp, Pp, Pp);
    launch_gemm(ah, wo1t + l * WCC, bo1 + l * Cc, t, t, nullptr, M0c, Cc, Cc, 1.0f, stream);

    // ---- cross-attention ----
    layernorm_kernel<<<dim3(M0c), 256, 0, stream>>>(t, ln2_g + l * Cc, ln2_b + l * Cc, tn);
    launch_gemm(tn,    wq2t + l * WCC, bq2 + l * Cc, nullptr, nullptr, qh, M0c, Cc, Cc, SCALEc, stream);
    launch_gemm(condh, wk2t + l * WCC, bk2 + l * Cc, nullptr, nullptr, kh, MTc, Cc, Cc, 1.0f, stream);
    launch_gemm(condh, wv2t + l * WCC, bv2 + l * Cc, nullptr, nullptr, vh, MTc, Cc, Cc, 1.0f, stream);
    vtrans_kernel<<<dim3(vtb_cross), 256, 0, stream>>>(vh, vt, Tt, SKPX);
    attn_wmma_kernel<<<attn_grid, 256, 0, stream>>>(qh, kh, vt, ah, Pp, Tt, SKPX);
    launch_gemm(ah, wo2t + l * WCC, bo2 + l * Cc, t, t, nullptr, M0c, Cc, Cc, 1.0f, stream);

    // ---- FFN (no activation, faithful) ----
    layernorm_kernel<<<dim3(M0c), 256, 0, stream>>>(t, ln3_g + l * Cc, ln3_b + l * Cc, tn);
    launch_gemm(tn,  fw1t + (size_t)l * Ff * Cc, fb1 + l * Ff, nullptr, nullptr, mid, M0c, Ff, Cc, 1.0f, stream);
    launch_gemm(mid, fw2t + (size_t)l * Cc * Ff, fb2 + l * Cc, t, t, nullptr, M0c, Cc, Ff, 1.0f, stream);
  }

  // Output projection + residual with original x
  f32tof16_kernel<<<dim3((unsigned)(((size_t)M0c * Cc + 255) / 256)), 256, 0, stream>>>(
      t, qh, (size_t)M0c * Cc);
  launch_gemm(qh, wout_t, wout_b, nullptr, proj, nullptr, M0c, Cc, Cc, 1.0f, stream);
  final_add_kernel<<<dim3((unsigned)(((size_t)Bn * Cc * Pp + 255) / 256)), 256, 0, stream>>>(
      proj, x, (float*)d_out);
}
